// UncertaintyAwareScoreDecoder_48808008351904
// MI455X (gfx1250) — compile-verified, matrix operand order checked by reading, not run on hardware
//
#include <hip/hip_runtime.h>
#include <math.h>

// ---------------------------------------------------------------------------
// Problem constants (match reference)
// ---------------------------------------------------------------------------
#define D_    768
#define C_    384
#define DC_   1152      // D + C
#define HW_   576       // 24*24
#define NB_   32        // batch
#define NG_   3         // memo groups
#define NT_   128       // NB + NG*NB samples through shared MAtt
#define TT_   144       // 12*12 template tokens
#define FF_   2304      // FFN hidden
#define SCALE_ 0.03608439182435161f   // 768^-0.5
#define EPS_  1e-5f

typedef __bf16 bf16_t;
typedef __attribute__((ext_vector_type(16))) __bf16 v16bf;
typedef __attribute__((ext_vector_type(8)))  float  v8f;

static __device__ __forceinline__ bf16_t f2bf(float f) {
  union { float f; unsigned u; } in; in.f = f;
  unsigned r = in.u + 0x7FFFu + ((in.u >> 16) & 1u);   // round-to-nearest-even
  union { unsigned short s; bf16_t b; } out; out.s = (unsigned short)(r >> 16);
  return out.b;
}

static __device__ __forceinline__ float waveReduceSum(float v) {
#pragma unroll
  for (int off = 16; off > 0; off >>= 1) v += __shfl_down(v, off, 32);
  return v;
}

// Map shared-path sample s in [0,128) to source index in the memo arrays.
// s < 32: search sample s.  s >= 32: s = 32 + b*G + g  ->  source row g*B + b
static __device__ __forceinline__ int memoSrc(int s) {
  int b = (s - NB_) / NG_, g = (s - NB_) % NG_;
  return g * NB_ + b;
}

// ---------------------------------------------------------------------------
// Tiled WMMA GEMM:  C = act(A[M,K] @ B + bias [+ residual])
//   transB=0: B is [K,N] row-major;  transB=1: B is [N,K] row-major (i.e. B^T)
// REQUIRES: K % 64 == 0, N % 64 == 0 (true for all GEMMs in this pipeline).
// Block = 256 threads = 8 waves arranged 4(M) x 2(N); block tile 64x64;
// K staged 64 at a time -> 4 v_wmma_f32_16x16x32_bf16 per barrier pair
// (each wave: 2 column tiles sharing one A fragment, 2 K substeps).
// Per-lane fragment layout per CDNA5 ISA 7.12.2 (16-bit A 16x32; f32 C/D 16x16).
// ---------------------------------------------------------------------------
__global__ __launch_bounds__(256)
void wmma_gemm_kernel(const float* __restrict__ A, const float* __restrict__ B,
                      const float* __restrict__ bias, const float* __restrict__ residual,
                      float* __restrict__ Cout, int M, int N, int K,
                      int transB, int relu)
{
  __shared__ bf16_t As[64][72];   // [m][k], row stride 144B (16B aligned)
  __shared__ bf16_t Bs[64][72];   // [n][k] (K-contiguous for b128 gathers)

  const int tid  = threadIdx.x;
  const int lane = tid & 31;
  const int wave = tid >> 5;
  const int wm   = wave >> 1;     // 0..3
  const int wn   = wave & 1;      // 0..1
  const int half = lane >> 4;     // ISA lane-half select
  const int mr   = lane & 15;
  const int rowBase = blockIdx.y * 64;
  const int colBase = blockIdx.x * 64;

  v8f acc0 = {0.f, 0.f, 0.f, 0.f, 0.f, 0.f, 0.f, 0.f};
  v8f acc1 = {0.f, 0.f, 0.f, 0.f, 0.f, 0.f, 0.f, 0.f};

  for (int kk = 0; kk < K; kk += 64) {
    // ---- stage A tile 64x64 (f32 -> bf16): 4 float4 per thread ----
#pragma unroll
    for (int it = 0; it < 4; ++it) {
      int i  = tid + it * 256;
      int r  = i >> 4;               // 16 quads per 64-wide row
      int c4 = (i & 15) << 2;
      int row = rowBase + r;
      float4 v = make_float4(0.f, 0.f, 0.f, 0.f);
      if (row < M) v = *(const float4*)(A + (size_t)row * K + kk + c4);
      As[r][c4 + 0] = f2bf(v.x); As[r][c4 + 1] = f2bf(v.y);
      As[r][c4 + 2] = f2bf(v.z); As[r][c4 + 3] = f2bf(v.w);
    }
    // ---- stage B tile 64(K) x 64(N) into Bs[n][k]: 4 float4 per thread ----
#pragma unroll
    for (int it = 0; it < 4; ++it) {
      int i  = tid + it * 256;
      int n  = i >> 4;
      int k4 = (i & 15) << 2;
      int col = colBase + n;         // N % 64 == 0 -> always in range
      float a0, a1, a2, a3;
      if (transB) {
        float4 v = *(const float4*)(B + (size_t)col * K + kk + k4);
        a0 = v.x; a1 = v.y; a2 = v.z; a3 = v.w;
      } else {
        a0 = B[(size_t)(kk + k4 + 0) * N + col];
        a1 = B[(size_t)(kk + k4 + 1) * N + col];
        a2 = B[(size_t)(kk + k4 + 2) * N + col];
        a3 = B[(size_t)(kk + k4 + 3) * N + col];
      }
      Bs[n][k4 + 0] = f2bf(a0); Bs[n][k4 + 1] = f2bf(a1);
      Bs[n][k4 + 2] = f2bf(a2); Bs[n][k4 + 3] = f2bf(a3);
    }
    __syncthreads();

    // ---- 2 K-substeps x 2 column tiles = 4 WMMAs per stage ----
    const int ar  = wm * 16 + mr;
    const int bn0 = wn * 16 + mr;
    const int bn1 = bn0 + 32;
#pragma unroll
    for (int ks = 0; ks < 2; ++ks) {
      const int kb = ks * 32;
      union { v16bf v; bf16_t e[16]; } af, bf0, bf1;
#pragma unroll
      for (int i = 0; i < 8; i++) {
        af.e[i]      = As[ar][kb + half * 8 + i];
        af.e[8 + i]  = As[ar][kb + 16 + half * 8 + i];
        bf0.e[i]     = Bs[bn0][kb + half * 8 + i];
        bf0.e[8 + i] = Bs[bn0][kb + 16 + half * 8 + i];
        bf1.e[i]     = Bs[bn1][kb + half * 8 + i];
        bf1.e[8 + i] = Bs[bn1][kb + 16 + half * 8 + i];
      }
      acc0 = __builtin_amdgcn_wmma_f32_16x16x32_bf16(false, af.v, false, bf0.v,
                                                     (short)0, acc0, false, false);
      acc1 = __builtin_amdgcn_wmma_f32_16x16x32_bf16(false, af.v, false, bf1.v,
                                                     (short)0, acc1, false, false);
    }
    __syncthreads();
  }

  // ---- store D (VGPR r -> M = r + 8*half; N = lane&15) ----
  const int col0 = colBase + wn * 16 + mr;
  const int col1 = col0 + 32;
  const float bv0 = bias ? bias[col0] : 0.f;
  const float bv1 = bias ? bias[col1] : 0.f;
#pragma unroll
  for (int r = 0; r < 8; r++) {
    int row = rowBase + wm * 16 + half * 8 + r;
    if (row < M) {
      float v0 = acc0[r] + bv0;
      float v1 = acc1[r] + bv1;
      if (residual) {
        v0 += residual[(size_t)row * N + col0];
        v1 += residual[(size_t)row * N + col1];
      }
      if (relu) { v0 = fmaxf(v0, 0.f); v1 = fmaxf(v1, 0.f); }
      Cout[(size_t)row * N + col0] = v0;
      Cout[(size_t)row * N + col1] = v1;
    }
  }
}

// ---------------------------------------------------------------------------
// GAP over template feature maps -> P0[128,768] (with memo (g,b)->(b,g) reorder)
// one wave per (s,d) row of 144 contiguous floats
// ---------------------------------------------------------------------------
__global__ __launch_bounds__(256)
void gap_kernel(const float* __restrict__ tf, const float* __restrict__ mtf,
                float* __restrict__ P0)
{
  int row  = blockIdx.x * 8 + (threadIdx.x >> 5);   // (s,d) flat, 0..98303
  int lane = threadIdx.x & 31;
  int s = row / D_, d = row % D_;
  const float* src = (s < NB_)
      ? tf  + ((size_t)s * D_ + d) * TT_
      : mtf + ((size_t)memoSrc(s) * D_ + d) * TT_;
  float acc = 0.f;
  for (int t = lane; t < TT_; t += 32) acc += src[t];
  acc = waveReduceSum(acc);
  if (lane == 0) P0[row] = acc * (1.f / TT_);
}

// ---------------------------------------------------------------------------
// Box -> {0,1} spatial mask [128,576]
// ---------------------------------------------------------------------------
__global__ __launch_bounds__(192)
void mask_kernel(const float* __restrict__ sbox, const float* __restrict__ mbox,
                 float* __restrict__ mask)
{
  int s = blockIdx.x;
  const float* bx = (s < NB_) ? sbox + s * 4 : mbox + memoSrc(s) * 4;
  float x0 = floorf(fminf(fmaxf(bx[0], 0.f), 1.f) * 24.f);
  float y0 = floorf(fminf(fmaxf(bx[1], 0.f), 1.f) * 24.f);
  float x1 = ceilf (fminf(fmaxf(bx[2], 0.f), 1.f) * 24.f);
  float y1 = ceilf (fminf(fmaxf(bx[3], 0.f), 1.f) * 24.f);
  for (int j = threadIdx.x; j < HW_; j += blockDim.x) {
    float r = (float)(j / 24), c = (float)(j % 24);
    mask[s * HW_ + j] = (r >= y0 && r < y1 && c >= x0 && c < x1) ? 1.f : 0.f;
  }
}

// ---------------------------------------------------------------------------
// bconst[s] = dot(QK[s], embd_b)   (the be·(wk q) attention bias term)
// ---------------------------------------------------------------------------
__global__ __launch_bounds__(256)
void dotb_kernel(const float* __restrict__ QK, const float* __restrict__ eb,
                 float* __restrict__ bconst)
{
  int s = blockIdx.x, t = threadIdx.x;
  __shared__ float red[256];
  float a = 0.f;
  for (int d = t; d < D_; d += 256) a += QK[(size_t)s * D_ + d] * eb[d];
  red[t] = a; __syncthreads();
  for (int off = 128; off > 0; off >>= 1) {
    if (t < off) red[t] += red[t + off];
    __syncthreads();
  }
  if (t == 0) bconst[s] = red[0];
}

// ---------------------------------------------------------------------------
// Streaming pass #1: logits_j = (X_j . WQK[s] + bconst)*SCALE + (1-mask)*-1e4
// then block softmax -> attn[s, 576].  Fully coalesced over spatial dim.
// ---------------------------------------------------------------------------
__global__ __launch_bounds__(576)
void attn_kernel(const float* __restrict__ sf,  const float* __restrict__ cm,
                 const float* __restrict__ msf, const float* __restrict__ mcm,
                 const float* __restrict__ WQK, const float* __restrict__ mask,
                 const float* __restrict__ bconst, float* __restrict__ attn)
{
  int s = blockIdx.x, j = threadIdx.x;
  const float *f, *c;
  if (s < NB_) { f = sf + (size_t)s * D_ * HW_;  c = cm + (size_t)s * C_ * HW_; }
  else { int src = memoSrc(s); f = msf + (size_t)src * D_ * HW_; c = mcm + (size_t)src * C_ * HW_; }

  __shared__ float w[DC_];
  for (int i = j; i < DC_; i += 576) w[i] = WQK[(size_t)s * DC_ + i];
  __syncthreads();

  float acc = 0.f;
  for (int ch = 0; ch < D_; ch++) acc += f[(size_t)ch * HW_ + j] * w[ch];
  for (int ch = 0; ch < C_; ch++) acc += c[(size_t)ch * HW_ + j] * w[D_ + ch];
  float logit = (acc + bconst[s]) * SCALE_ + (1.f - mask[s * HW_ + j]) * (-10000.f);

  __shared__ float red[576];
  red[j] = logit; __syncthreads();
  if (j < 64) red[j] = fmaxf(red[j], red[j + 512]);
  __syncthreads();
  for (int off = 256; off > 0; off >>= 1) {
    if (j < off) red[j] = fmaxf(red[j], red[j + off]);
    __syncthreads();
  }
  float mx = red[0]; __syncthreads();
  float e = expf(logit - mx);
  red[j] = e; __syncthreads();
  if (j < 64) red[j] += red[j + 512];
  __syncthreads();
  for (int off = 256; off > 0; off >>= 1) {
    if (j < off) red[j] += red[j + off];
    __syncthreads();
  }
  attn[(size_t)s * HW_ + j] = e / red[0];
}

// ---------------------------------------------------------------------------
// Streaming pass #2: AXP[s,c] = sum_j attn[s,j] * X[s,c,j]  (1152 channels)
// one wave per channel; 32-lane coalesced reads of the feature maps
// ---------------------------------------------------------------------------
__global__ __launch_bounds__(256)
void axp_kernel(const float* __restrict__ sf,  const float* __restrict__ cm,
                const float* __restrict__ msf, const float* __restrict__ mcm,
                const float* __restrict__ attn, float* __restrict__ AXP)
{
  int s = blockIdx.y;
  const float *f, *c;
  if (s < NB_) { f = sf + (size_t)s * D_ * HW_;  c = cm + (size_t)s * C_ * HW_; }
  else { int src = memoSrc(s); f = msf + (size_t)src * D_ * HW_; c = mcm + (size_t)src * C_ * HW_; }

  __shared__ float a[HW_];
  for (int i = threadIdx.x; i < HW_; i += 256) a[i] = attn[(size_t)s * HW_ + i];
  __syncthreads();

  int wave = threadIdx.x >> 5, lane = threadIdx.x & 31;
  int ch = blockIdx.x * 8 + wave;                    // 0..1151
  const float* X = (ch < D_) ? f + (size_t)ch * HW_ : c + (size_t)(ch - D_) * HW_;
  float acc = 0.f;
  for (int j = lane; j < HW_; j += 32) acc += X[j] * a[j];
  acc = waveReduceSum(acc);
  if (lane == 0) AXP[(size_t)s * DC_ + ch] = acc;
}

// ---------------------------------------------------------------------------
// Row LayerNorm over D=768
// ---------------------------------------------------------------------------
__global__ __launch_bounds__(256)
void ln_kernel(const float* __restrict__ X, const float* __restrict__ g,
               const float* __restrict__ b, float* __restrict__ Y)
{
  int row = blockIdx.x, t = threadIdx.x;
  __shared__ float r1[256], r2[256];
  float s1 = 0.f, s2 = 0.f;
  for (int d = t; d < D_; d += 256) {
    float v = X[(size_t)row * D_ + d]; s1 += v; s2 += v * v;
  }
  r1[t] = s1; r2[t] = s2; __syncthreads();
  for (int off = 128; off > 0; off >>= 1) {
    if (t < off) { r1[t] += r1[t + off]; r2[t] += r2[t + off]; }
    __syncthreads();
  }
  float mean = r1[0] * (1.f / D_);
  float var  = r2[0] * (1.f / D_) - mean * mean;
  float rstd = rsqrtf(var + EPS_);
  for (int d = t; d < D_; d += 256)
    Y[(size_t)row * D_ + d] = (X[(size_t)row * D_ + d] - mean) * rstd * g[d] + b[d];
}

// ---------------------------------------------------------------------------
// proto2memo attention core: 3 keys per batch item
// ctxv[b] = sum_g softmax_g(Qm[b].Km[b,g]*SCALE) * Vm[b,g]
// ---------------------------------------------------------------------------
__global__ __launch_bounds__(96)
void memo_attn_kernel(const float* __restrict__ Qm, const float* __restrict__ Km,
                      const float* __restrict__ Vm, float* __restrict__ ctxv)
{
  int b = blockIdx.x;
  int wave = threadIdx.x >> 5, lane = threadIdx.x & 31;
  float acc = 0.f;
  for (int d = lane; d < D_; d += 32)
    acc += Qm[(size_t)b * D_ + d] * Km[(size_t)(b * 3 + wave) * D_ + d];
  acc = waveReduceSum(acc);
  __shared__ float sc[3];
  if (lane == 0) sc[wave] = acc * SCALE_;
  __syncthreads();
  float m  = fmaxf(sc[0], fmaxf(sc[1], sc[2]));
  float e0 = expf(sc[0] - m), e1 = expf(sc[1] - m), e2 = expf(sc[2] - m);
  float inv = 1.f / (e0 + e1 + e2);
  float a0 = e0 * inv, a1 = e1 * inv, a2 = e2 * inv;
  for (int d = threadIdx.x; d < D_; d += 96)
    ctxv[(size_t)b * D_ + d] = a0 * Vm[(size_t)(b * 3 + 0) * D_ + d]
                             + a1 * Vm[(size_t)(b * 3 + 1) * D_ + d]
                             + a2 * Vm[(size_t)(b * 3 + 2) * D_ + d];
}

// ---------------------------------------------------------------------------
// proto2template attention core: 144 keys, scores via qkt = wk @ q folding
// ctxt[b,d] = sum_t softmax_t(tmpl[b,t].qkt[b]*SCALE) * tf[b,d,t]
// ---------------------------------------------------------------------------
__global__ __launch_bounds__(256)
void tmpl_attn_kernel(const float* __restrict__ tf, const float* __restrict__ qkt,
                      float* __restrict__ ctxt)
{
  int b = blockIdx.x, t = threadIdx.x;
  __shared__ float q[D_];
  __shared__ float sc[TT_];
  __shared__ float an[TT_];
  for (int d = t; d < D_; d += 256) q[d] = qkt[(size_t)b * D_ + d];
  __syncthreads();
  if (t < TT_) {
    float acc = 0.f;
    const float* base = tf + (size_t)b * D_ * TT_ + t;
    for (int d = 0; d < D_; d++) acc += base[(size_t)d * TT_] * q[d];
    sc[t] = acc * SCALE_;
  }
  __syncthreads();
  // everyone scans the 144 logits (cheap), lanes t<144 publish normalized attn
  float m = -1e30f;
  for (int i = 0; i < TT_; i++) m = fmaxf(m, sc[i]);
  float sum = 0.f;
  for (int i = 0; i < TT_; i++) sum += expf(sc[i] - m);
  float inv = 1.f / sum;
  if (t < TT_) an[t] = expf(sc[t] - m) * inv;
  __syncthreads();
  for (int d = t; d < D_; d += 256) {
    const float* row = tf + ((size_t)b * D_ + d) * TT_;
    float acc = 0.f;
    for (int i = 0; i < TT_; i++) acc += row[i] * an[i];
    ctxt[(size_t)b * D_ + d] = acc;
  }
}

// ---------------------------------------------------------------------------
// Head GEMV: out[b] = H1[b,:] . mlp_w2 + mlp_b2   (N=1, one wave per row)
// ---------------------------------------------------------------------------
__global__ __launch_bounds__(256)
void head_kernel(const float* __restrict__ H1, const float* __restrict__ w2,
                 const float* __restrict__ b2, float* __restrict__ out)
{
  int b = blockIdx.x * 8 + (threadIdx.x >> 5);
  int lane = threadIdx.x & 31;
  float acc = 0.f;
  for (int d = lane; d < D_; d += 32) acc += H1[(size_t)b * D_ + d] * w2[d];
  acc = waveReduceSum(acc);
  if (lane == 0) out[b] = acc + b2[0];
}

// ---------------------------------------------------------------------------
// Launcher
// ---------------------------------------------------------------------------
extern "C" void kernel_launch(void* const* d_in, const int* in_sizes, int n_in,
                              void* d_out, int out_size, void* d_ws, size_t ws_size,
                              hipStream_t stream) {
  (void)in_sizes; (void)n_in; (void)out_size; (void)ws_size;
  // inputs (setup_inputs dict order; params flattened in insertion order)
  const float* search_feat   = (const float*)d_in[0];
  const float* template_feat = (const float*)d_in[1];
  const float* search_box    = (const float*)d_in[2];
  const float* conf_map      = (const float*)d_in[3];
  const float* memo_coord    = (const float*)d_in[4];
  const float* mem_sfeat     = (const float*)d_in[5];
  const float* mem_cmap      = (const float*)d_in[6];
  const float* mem_tfeat     = (const float*)d_in[7];
  const float* embd_w  = (const float*)d_in[8];
  const float* embd_b  = (const float*)d_in[9];
  const float* s_wq    = (const float*)d_in[10];
  const float* s_wk    = (const float*)d_in[11];
  const float* s_wv    = (const float*)d_in[12];
  const float* s_wo    = (const float*)d_in[13];
  const float* s_ln1_g = (const float*)d_in[14];
  const float* s_ln1_b = (const float*)d_in[15];
  const float* s_ffn1  = (const float*)d_in[16];
  const float* s_ffn2  = (const float*)d_in[17];
  const float* s_ln2_g = (const float*)d_in[18];
  const float* s_ln2_b = (const float*)d_in[19];
  const float* m_wq = (const float*)d_in[20];
  const float* m_wk = (const float*)d_in[21];
  const float* m_wv = (const float*)d_in[22];
  const float* m_wo = (const float*)d_in[23];
  const float* t_wq = (const float*)d_in[24];
  const float* t_wk = (const float*)d_in[25];
  const float* t_wv = (const float*)d_in[26];
  const float* t_wo = (const float*)d_in[27];
  const float* mlp_w0 = (const float*)d_in[28];
  const float* mlp_b0 = (const float*)d_in[29];
  const float* mlp_w1 = (const float*)d_in[30];
  const float* mlp_b1 = (const float*)d_in[31];
  const float* mlp_w2 = (const float*)d_in[32];
  const float* mlp_b2 = (const float*)d_in[33];
  float* out = (float*)d_out;

  // workspace layout (floats)
  float* ws = (float*)d_ws;
  size_t off = 0;
  auto alloc = [&](size_t n) { float* p = ws + off; off += n; return p; };
  float* mask    = alloc((size_t)NT_ * HW_);
  float* P0      = alloc((size_t)NT_ * D_);
  float* Q       = alloc((size_t)NT_ * D_);
  float* QK      = alloc((size_t)NT_ * D_);
  float* WQK     = alloc((size_t)NT_ * DC_);
  float* bconst  = alloc(NT_);
  float* attn    = alloc((size_t)NT_ * HW_);
  float* AXP     = alloc((size_t)NT_ * DC_);
  float* CTX     = alloc((size_t)NT_ * D_);
  float* X1      = alloc((size_t)NT_ * D_);
  float* X1ln    = alloc((size_t)NT_ * D_);
  float* Hffn    = alloc((size_t)NT_ * FF_);
  float* X2      = alloc((size_t)NT_ * D_);
  float* protoA  = alloc((size_t)NT_ * D_);
  float* WVO     = alloc((size_t)D_ * D_);
  float* WVOT    = alloc((size_t)D_ * D_);
  float* Qm      = alloc((size_t)NB_ * D_);
  float* Km      = alloc((size_t)NB_ * NG_ * D_);
  float* Vm      = alloc((size_t)NB_ * NG_ * D_);
  float* ctxv    = alloc((size_t)NB_ * D_);
  float* P1      = alloc((size_t)NB_ * D_);
  float* Qt      = alloc((size_t)NB_ * D_);
  float* qkt     = alloc((size_t)NB_ * D_);
  float* ctxt    = alloc((size_t)NB_ * D_);
  float* P2      = alloc((size_t)NB_ * D_);
  float* H0      = alloc((size_t)NB_ * D_);
  float* H1      = alloc((size_t)NB_ * D_);

  auto gemm = [&](const float* A, const float* B, const float* bias,
                  const float* res, float* C, int M, int N, int K,
                  int transB, int relu) {
    dim3 g((N + 63) / 64, (M + 63) / 64);
    hipLaunchKernelGGL(wmma_gemm_kernel, g, dim3(256), 0, stream,
                       A, B, bias, res, C, M, N, K, transB, relu);
  };

  // ---- prototype generation (rank-1 attention, folded) ----
  hipLaunchKernelGGL(gap_kernel, dim3(NT_ * D_ / 8), dim3(256), 0, stream,
                     template_feat, mem_tfeat, P0);
  hipLaunchKernelGGL(mask_kernel, dim3(NT_), dim3(192), 0, stream,
                     search_box, memo_coord, mask);
  gemm(P0, s_wq, nullptr, nullptr, Q, NT_, D_, D_, 0, 0);          // q = P0 @ wq
  gemm(Q, s_wk, nullptr, nullptr, QK, NT_, D_, D_, 1, 0);          // wk @ q
  hipLaunchKernelGGL(dotb_kernel, dim3(NT_), dim3(256), 0, stream, QK, embd_b, bconst);
  gemm(QK, embd_w, nullptr, nullptr, WQK, NT_, DC_, D_, 1, 0);     // we @ (wk q)
  hipLaunchKernelGGL(attn_kernel, dim3(NT_), dim3(576), 0, stream,
                     search_feat, conf_map, mem_sfeat, mem_cmap,
                     WQK, mask, bconst, attn);
  hipLaunchKernelGGL(axp_kernel, dim3(DC_ / 8, NT_), dim3(256), 0, stream,
                     search_feat, conf_map, mem_sfeat, mem_cmap, attn, AXP);
  gemm(AXP, embd_w, embd_b, nullptr, CTX, NT_, D_, DC_, 0, 0);     // ctx = axp@we + be
  gemm(s_wv, s_wo, nullptr, nullptr, WVO, D_, D_, D_, 0, 0);       // fuse wv@wo
  gemm(CTX, WVO, nullptr, nullptr, X1, NT_, D_, D_, 0, 0);
  hipLaunchKernelGGL(ln_kernel, dim3(NT_), dim3(256), 0, stream, X1, s_ln1_g, s_ln1_b, X1ln);
  gemm(X1ln, s_ffn1, nullptr, nullptr, Hffn, NT_, FF_, D_, 0, 1);  // relu(x@w1)
  gemm(Hffn, s_ffn2, nullptr, X1ln, X2, NT_, D_, FF_, 0, 0);       // x + h@w2
  hipLaunchKernelGGL(ln_kernel, dim3(NT_), dim3(256), 0, stream, X2, s_ln2_g, s_ln2_b, protoA);

  // search_prototype output (rows 0..31 of protoA)
  hipMemcpyAsync(out + NB_, protoA, (size_t)NB_ * D_ * sizeof(float),
                 hipMemcpyDeviceToDevice, stream);

  // ---- proto2memo ----
  const float* memoP = protoA + (size_t)NB_ * D_;                  // [32*3, 768]
  gemm(protoA, m_wq, nullptr, nullptr, Qm, NB_, D_, D_, 0, 0);
  gemm(memoP, m_wk, nullptr, nullptr, Km, NB_ * NG_, D_, D_, 0, 0);
  gemm(memoP, m_wv, nullptr, nullptr, Vm, NB_ * NG_, D_, D_, 0, 0);
  hipLaunchKernelGGL(memo_attn_kernel, dim3(NB_), dim3(96), 0, stream, Qm, Km, Vm, ctxv);
  gemm(ctxv, m_wo, nullptr, protoA, P1, NB_, D_, D_, 0, 0);        // residual

  // ---- proto2template ----
  gemm(P1, t_wq, nullptr, nullptr, Qt, NB_, D_, D_, 0, 0);
  gemm(Qt, t_wk, nullptr, nullptr, qkt, NB_, D_, D_, 1, 0);        // wk @ q
  hipLaunchKernelGGL(tmpl_attn_kernel, dim3(NB_), dim3(256), 0, stream,
                     template_feat, qkt, ctxt);
  gemm(t_wv, t_wo, nullptr, nullptr, WVOT, D_, D_, D_, 0, 0);      // fuse wv@wo
  gemm(ctxt, WVOT, nullptr, P1, P2, NB_, D_, D_, 0, 0);            // residual

  // ---- out_score MLP ----
  gemm(P2, mlp_w0, mlp_b0, nullptr, H0, NB_, D_, D_, 0, 1);
  gemm(H0, mlp_w1, mlp_b1, nullptr, H1, NB_, D_, D_, 0, 1);
  hipLaunchKernelGGL(head_kernel, dim3(NB_ / 8), dim3(256), 0, stream,
                     H1, mlp_w2, mlp_b2, out);
}
